// SelfAttention_47983374631319
// MI455X (gfx1250) — matrix operand order repaired and checked by reading, not compiled, not run on hardware
//
#include <hip/hip_runtime.h>

// ---------------------------------------------------------------------------
// Self-attention for MI455X (gfx1250): bf16 WMMA, f32 accumulate,
// async global->LDS staging in the attention kernel, double-buffered GEMMs.
// ---------------------------------------------------------------------------

typedef __attribute__((ext_vector_type(16))) __bf16       v16bf;
typedef __attribute__((ext_vector_type(8)))  float        v8f;
typedef __attribute__((ext_vector_type(4)))  unsigned int u32x4;
typedef __attribute__((ext_vector_type(8)))  unsigned int u32x8;
typedef int v4i_gcc __attribute__((vector_size(16)));  // matches builtin param

#define D_HID   1024
#define N_SEQ   2048
#define N_BATCH 4
#define N_HEAD  16
#define D_HEAD  64

#define GLOBAL_AS __attribute__((address_space(1)))
#define LDS_AS    __attribute__((address_space(3)))

#if defined(__has_builtin)
#if __has_builtin(__builtin_amdgcn_global_load_async_to_lds_b128)
#define HAVE_ASYNC_CP 1
#endif
#if __has_builtin(__builtin_amdgcn_s_wait_asynccnt)
#define HAVE_WAIT_ASYNC 1
#endif
#endif

// round-to-nearest-even f32 -> bf16
static __device__ __forceinline__ unsigned short f2bf(float f) {
  unsigned int u = __builtin_bit_cast(unsigned int, f);
  u += 0x7FFFu + ((u >> 16) & 1u);
  return (unsigned short)(u >> 16);
}

// Load a 16-element bf16 fragment as two 16-byte chunks (global or LDS).
static __device__ __forceinline__ v16bf load_frag(const unsigned short* p0,
                                                  const unsigned short* p1) {
  u32x4 a = *(const u32x4*)p0;
  u32x4 b = *(const u32x4*)p1;
  u32x8 u = {a[0], a[1], a[2], a[3], b[0], b[1], b[2], b[3]};
  return __builtin_bit_cast(v16bf, u);
}

#define WMMA_BF16(A, B, C)                                                    \
  __builtin_amdgcn_wmma_f32_16x16x32_bf16(false, (A), false, (B), (short)0,   \
                                          (C), false, false)

// 16-byte global -> LDS copy: async DMA path on gfx1250, sync fallback.
static __device__ __forceinline__ void async_cp16(const unsigned short* g,
                                                  unsigned short* l) {
#ifdef HAVE_ASYNC_CP
  __builtin_amdgcn_global_load_async_to_lds_b128(
      (GLOBAL_AS v4i_gcc*)g, (LDS_AS v4i_gcc*)l, 0, 0);
#else
  *(u32x4*)l = *(const u32x4*)g;
#endif
}

static __device__ __forceinline__ void wait_async0() {
#ifdef HAVE_WAIT_ASYNC
  __builtin_amdgcn_s_wait_asynccnt(0);
#else
  asm volatile("s_wait_asynccnt 0x0" ::: "memory");
#endif
}

// ---------------------------------------------------------------------------
// Kernel 0: fp32 -> bf16 conversion (4 elements / thread)
// ---------------------------------------------------------------------------
__global__ __launch_bounds__(256) void cvt_bf16(const float* __restrict__ src,
                                                unsigned short* __restrict__ dst,
                                                int n4) {
  int i = blockIdx.x * blockDim.x + threadIdx.x;
  if (i >= n4) return;
  float4 v = ((const float4*)src)[i];
  unsigned int lo = (unsigned int)f2bf(v.x) | ((unsigned int)f2bf(v.y) << 16);
  unsigned int hi = (unsigned int)f2bf(v.z) | ((unsigned int)f2bf(v.w) << 16);
  ((uint2*)dst)[i] = make_uint2(lo, hi);
}

// ---------------------------------------------------------------------------
// Double-buffered bf16 GEMM core: one wave computes a 32x64 C tile.
//   C[32,64] += A[32,K] * B[64,K]^T, K = 1024, both operands row-major [*,K].
// ---------------------------------------------------------------------------
static __device__ __forceinline__ void gemm_load(v16bf a[2], v16bf b[4],
                                                 const unsigned short* aRow0,
                                                 const unsigned short* aRow1,
                                                 const unsigned short* const bCol[4],
                                                 int k0, int akb, int bkb) {
  a[0] = load_frag(aRow0 + k0 + akb, aRow0 + k0 + akb + 16);
  a[1] = load_frag(aRow1 + k0 + akb, aRow1 + k0 + akb + 16);
#pragma unroll
  for (int nn = 0; nn < 4; ++nn)
    b[nn] = load_frag(bCol[nn] + k0 + bkb, bCol[nn] + k0 + bkb + 16);
}

static __device__ __forceinline__ void gemm_mma(v8f acc[2][4], const v16bf a[2],
                                                const v16bf b[4]) {
#pragma unroll
  for (int nn = 0; nn < 4; ++nn) {
    acc[0][nn] = WMMA_BF16(a[0], b[nn], acc[0][nn]);
    acc[1][nn] = WMMA_BF16(a[1], b[nn], acc[1][nn]);
  }
}

static __device__ __forceinline__ void gemm_core(v8f acc[2][4],
                                                 const unsigned short* aRow0,
                                                 const unsigned short* aRow1,
                                                 const unsigned short* const bCol[4],
                                                 int akb, int bkb) {
  v16bf a0[2], b0[4], a1[2], b1[4];
  gemm_load(a0, b0, aRow0, aRow1, bCol, 0, akb, bkb);
  for (int k0 = 0; k0 < D_HID; k0 += 64) {
    __builtin_prefetch(aRow0 + k0 + 512, 0, 1);
    gemm_load(a1, b1, aRow0, aRow1, bCol, k0 + 32, akb, bkb);   // next half
    gemm_mma(acc, a0, b0);                                      // overlap
    gemm_load(a0, b0, aRow0, aRow1, bCol, (k0 + 64) & (D_HID - 1), akb, bkb);
    gemm_mma(acc, a1, b1);
  }
}

// ---------------------------------------------------------------------------
// Kernel 1: QKV GEMM. C[8192,3072] = x_bf16 * qkv_w_bf16^T + bias.
// Block = 256 (8 waves); wave tile 32x64 -> block tile 256x64.
// Epilogue scatters into Q,K ([bh,n,64]) and V^T ([bh,64,n]).
// ---------------------------------------------------------------------------
__global__ __launch_bounds__(256) void qkv_gemm(
    const unsigned short* __restrict__ xb,    // [8192,1024]
    const unsigned short* __restrict__ wb,    // [3072,1024]
    const float* __restrict__ bias,           // [3072]
    unsigned short* __restrict__ qbuf,        // [64,2048,64]
    unsigned short* __restrict__ kbuf,        // [64,2048,64]
    unsigned short* __restrict__ vtbuf) {     // [64,64,2048]
  const int lane  = threadIdx.x & 31;
  const int wave  = threadIdx.x >> 5;
  const int col   = lane & 15;
  const int hi    = lane >> 4;
  const int mBase = blockIdx.x * 256 + wave * 32;
  const int eBase = blockIdx.y * 64;
  const int akb   = hi * 8;
  const int bkb   = hi * 8;

  const unsigned short* aRow0 = xb + (size_t)(mBase + col) * D_HID;
  const unsigned short* aRow1 = aRow0 + (size_t)16 * D_HID;
  const unsigned short* bCol[4];
#pragma unroll
  for (int nn = 0; nn < 4; ++nn)
    bCol[nn] = wb + (size_t)(eBase + nn * 16 + col) * D_HID;

  v8f acc[2][4] = {};
  gemm_core(acc, aRow0, aRow1, bCol, akb, bkb);

  const int which = eBase >> 10;          // 0=q 1=k 2=v (uniform per block)
  const int head  = (eBase & 1023) >> 6;  // head index (uniform per block)
  float bv[4];
#pragma unroll
  for (int nn = 0; nn < 4; ++nn) bv[nn] = bias[eBase + nn * 16 + col];

#pragma unroll
  for (int rh = 0; rh < 2; ++rh)
#pragma unroll
    for (int nn = 0; nn < 4; ++nn) {
      const int dh = nn * 16 + col;
#pragma unroll
      for (int r = 0; r < 8; ++r) {
        const int m  = mBase + rh * 16 + r + 8 * hi;
        const int b  = m >> 11;   // batch
        const int ns = m & 2047;  // sequence position
        const int bh = b * N_HEAD + head;
        const unsigned short v = f2bf(acc[rh][nn][r] + bv[nn]);
        if (which == 0)
          qbuf[((size_t)bh * N_SEQ + ns) * D_HEAD + dh] = v;
        else if (which == 1)
          kbuf[((size_t)bh * N_SEQ + ns) * D_HEAD + dh] = v;
        else
          vtbuf[((size_t)bh * D_HEAD + dh) * N_SEQ + ns] = v;
      }
    }
}

// ---------------------------------------------------------------------------
// Kernel 2: flash attention. Block = 128 (4 waves) shares one (batch,head);
// K/V^T 32-key tiles are async-staged into double-buffered LDS once per block;
// each wave owns 16 queries with online softmax.
// ---------------------------------------------------------------------------

// Stage one 32-key K tile (32x64 bf16) + V^T tile (64x32 bf16) into LDS.
// 128 threads x 2 chunks x 16B per tile.
static __device__ __forceinline__ void stage_tiles(
    const unsigned short* __restrict__ kg,   // kbuf + bh*2048*64
    const unsigned short* __restrict__ vg,   // vtbuf + bh*64*2048
    unsigned short* kl, unsigned short* vl, int kt, int t) {
#pragma unroll
  for (int c0 = 0; c0 < 2; ++c0) {
    const int c = t + c0 * 128;  // chunk 0..255 (16B each)
    async_cp16(kg + (size_t)(kt + (c >> 3)) * D_HEAD + (c & 7) * 8, kl + c * 8);
    async_cp16(vg + (size_t)(c >> 2) * N_SEQ + kt + (c & 3) * 8, vl + c * 8);
  }
}

__global__ __launch_bounds__(128) void attn(
    const unsigned short* __restrict__ qbuf,   // [64,2048,64]
    const unsigned short* __restrict__ kbuf,   // [64,2048,64]
    const unsigned short* __restrict__ vtbuf,  // [64,64,2048]
    unsigned short* __restrict__ ob) {         // [4,2048,1024] bf16
  __shared__ __align__(16) unsigned short kt_lds[2][32 * 64];  // 8 KiB
  __shared__ __align__(16) unsigned short vt_lds[2][64 * 32];  // 8 KiB
  __shared__ __align__(16) unsigned short pl[4][16 * 32];      // 4 KiB

  const int tid   = threadIdx.x;
  const int lane  = tid & 31;
  const int wave  = tid >> 5;
  const int col   = lane & 15;
  const int hi    = lane >> 4;
  const int bh    = blockIdx.y;
  const int qBase = blockIdx.x * 64 + wave * 16;
  const int akb   = hi * 8;
  const int bkb   = hi * 8;
  const float scale = 0.125f;  // 1/sqrt(64)
  const float L2E   = 1.44269504f;

  const unsigned short* kg = kbuf + (size_t)bh * N_SEQ * D_HEAD;
  const unsigned short* vg = vtbuf + (size_t)bh * D_HEAD * N_SEQ;

  // Q fragments held in registers for the whole kernel (A layout).
  const unsigned short* qRow = qbuf + ((size_t)bh * N_SEQ + qBase + col) * D_HEAD;
  v16bf qa0 = load_frag(qRow + akb, qRow + akb + 16);
  v16bf qa1 = load_frag(qRow + 32 + akb, qRow + 32 + akb + 16);

  v8f o0 = {}, o1 = {}, o2 = {}, o3 = {};
  float mrow[8], lrow[8];
#pragma unroll
  for (int r = 0; r < 8; ++r) { mrow[r] = -3.0e38f; lrow[r] = 0.0f; }

  unsigned short* myp = pl[wave];

  stage_tiles(kg, vg, kt_lds[0], vt_lds[0], 0, tid);  // prologue stage

  for (int kt = 0; kt < N_SEQ; kt += 32) {
    const int cur = (kt >> 5) & 1;
    const unsigned short* kl = kt_lds[cur];
    const unsigned short* vl = vt_lds[cur];

    wait_async0();     // my part of buffer `cur` landed in LDS
    __syncthreads();   // everyone's part landed; prior reads of !cur finished
    if (kt + 32 < N_SEQ)
      stage_tiles(kg, vg, kt_lds[cur ^ 1], vt_lds[cur ^ 1], kt + 32, tid);

    // ---- S = scale * Q K^T for 32 keys (two 16x16 tiles), K frags from LDS -
    const unsigned short* kr0 = kl + (size_t)col * D_HEAD;
    const unsigned short* kr1 = kr0 + 16 * D_HEAD;
    v8f s0 = {}, s1 = {};
    {
      v16bf b;
      b = load_frag(kr0 + bkb, kr0 + bkb + 16);            s0 = WMMA_BF16(qa0, b, s0);
      b = load_frag(kr0 + 32 + bkb, kr0 + 32 + bkb + 16);  s0 = WMMA_BF16(qa1, b, s0);
      b = load_frag(kr1 + bkb, kr1 + bkb + 16);            s1 = WMMA_BF16(qa0, b, s1);
      b = load_frag(kr1 + 32 + bkb, kr1 + 32 + bkb + 16);  s1 = WMMA_BF16(qa1, b, s1);
    }

    // ---- online softmax (row stats via cross-lane butterfly) ---------------
    float rmax[8];
#pragma unroll
    for (int r = 0; r < 8; ++r) {
      s0[r] *= scale; s1[r] *= scale;
      rmax[r] = fmaxf(s0[r], s1[r]);
    }
#pragma unroll
    for (int mask = 1; mask <= 8; mask <<= 1)
#pragma unroll
      for (int r = 0; r < 8; ++r)
        rmax[r] = fmaxf(rmax[r], __shfl_xor(rmax[r], mask, 32));

    float corr[8], p0v[8], p1v[8], rsum[8];
#pragma unroll
    for (int r = 0; r < 8; ++r) {
      float mn = fmaxf(mrow[r], rmax[r]);
      corr[r]  = exp2f((mrow[r] - mn) * L2E);
      mrow[r]  = mn;
      p0v[r]   = exp2f((s0[r] - mn) * L2E);
      p1v[r]   = exp2f((s1[r] - mn) * L2E);
      rsum[r]  = p0v[r] + p1v[r];
    }
#pragma unroll
    for (int mask = 1; mask <= 8; mask <<= 1)
#pragma unroll
      for (int r = 0; r < 8; ++r)
        rsum[r] += __shfl_xor(rsum[r], mask, 32);
#pragma unroll
    for (int r = 0; r < 8; ++r) lrow[r] = lrow[r] * corr[r] + rsum[r];

    // ---- P: C-layout -> per-wave LDS tile (bf16) -> A-layout fragment ------
#pragma unroll
    for (int r = 0; r < 8; ++r) {
      const int m = r + 8 * hi;
      myp[m * 32 + col]      = f2bf(p0v[r]);
      myp[m * 32 + 16 + col] = f2bf(p1v[r]);
    }
    asm volatile("s_wait_dscnt 0x0" ::: "memory");  // per-wave LDS RAW
    const unsigned short* pr = myp + col * 32;
    v16bf pa = load_frag(pr + akb, pr + akb + 16);
    asm volatile("" ::: "memory");

    // ---- O = corr*O + P V, V frags from LDS --------------------------------
#pragma unroll
    for (int r = 0; r < 8; ++r) {
      o0[r] *= corr[r]; o1[r] *= corr[r]; o2[r] *= corr[r]; o3[r] *= corr[r];
    }
    const unsigned short* vr = vl + (size_t)col * 32 + bkb;
    {
      v16bf b;
      b = load_frag(vr,            vr + 16);            o0 = WMMA_BF16(pa, b, o0);
      b = load_frag(vr + 16 * 32,  vr + 16 * 32 + 16);  o1 = WMMA_BF16(pa, b, o1);
      b = load_frag(vr + 32 * 32,  vr + 32 * 32 + 16);  o2 = WMMA_BF16(pa, b, o2);
      b = load_frag(vr + 48 * 32,  vr + 48 * 32 + 16);  o3 = WMMA_BF16(pa, b, o3);
    }
  }

  // ---- finalize: O / l, write bf16 [b, n, h*64 + dh] -----------------------
  const int bb = bh >> 4, head = bh & 15;
#pragma unroll
  for (int r = 0; r < 8; ++r) {
    const float inv = 1.0f / lrow[r];
    const int q = qBase + r + 8 * hi;
    unsigned short* orow = ob + ((size_t)(bb * N_SEQ + q)) * D_HID + head * D_HEAD;
    orow[0 * 16 + col] = f2bf(o0[r] * inv);
    orow[1 * 16 + col] = f2bf(o1[r] * inv);
    orow[2 * 16 + col] = f2bf(o2[r] * inv);
    orow[3 * 16 + col] = f2bf(o3[r] * inv);
  }
}

// ---------------------------------------------------------------------------
// Kernel 3: output projection. out[8192,1024] = attn_bf16 * o_w^T + o_b (f32)
// ---------------------------------------------------------------------------
__global__ __launch_bounds__(256) void out_gemm(
    const unsigned short* __restrict__ ab,   // [8192,1024] bf16
    const unsigned short* __restrict__ wb,   // [1024,1024] bf16
    const float* __restrict__ bias,          // [1024]
    float* __restrict__ out) {               // [8192,1024] f32
  const int lane  = threadIdx.x & 31;
  const int wave  = threadIdx.x >> 5;
  const int col   = lane & 15;
  const int hi    = lane >> 4;
  const int mBase = blockIdx.x * 256 + wave * 32;
  const int nBase = blockIdx.y * 64;
  const int akb   = hi * 8;
  const int bkb   = hi * 8;

  const unsigned short* aRow0 = ab + (size_t)(mBase + col) * D_HID;
  const unsigned short* aRow1 = aRow0 + (size_t)16 * D_HID;
  const unsigned short* bCol[4];
#pragma unroll
  for (int nn = 0; nn < 4; ++nn)
    bCol[nn] = wb + (size_t)(nBase + nn * 16 + col) * D_HID;

  v8f acc[2][4] = {};
  gemm_core(acc, aRow0, aRow1, bCol, akb, bkb);

  float bv[4];
#pragma unroll
  for (int nn = 0; nn < 4; ++nn) bv[nn] = bias[nBase + nn * 16 + col];
#pragma unroll
  for (int rh = 0; rh < 2; ++rh)
#pragma unroll
    for (int nn = 0; nn < 4; ++nn)
#pragma unroll
      for (int r = 0; r < 8; ++r) {
        const int m = mBase + rh * 16 + r + 8 * hi;
        out[(size_t)m * D_HID + nBase + nn * 16 + col] = acc[rh][nn][r] + bv[nn];
      }
}

// ---------------------------------------------------------------------------
// Host launcher
// ---------------------------------------------------------------------------
extern "C" void kernel_launch(void* const* d_in, const int* in_sizes, int n_in,
                              void* d_out, int out_size, void* d_ws,
                              size_t ws_size, hipStream_t stream) {
  const float* x     = (const float*)d_in[0];  // [4,2048,1024]
  const float* qkv_w = (const float*)d_in[1];  // [3072,1024]
  const float* qkv_b = (const float*)d_in[2];  // [3072]
  const float* o_w   = (const float*)d_in[3];  // [1024,1024]
  const float* o_b   = (const float*)d_in[4];  // [1024]
  float* out = (float*)d_out;

  char* ws = (char*)d_ws;
  unsigned short* xb   = (unsigned short*)(ws + 0);          // 16 MiB
  unsigned short* qwb  = (unsigned short*)(ws + 16777216);   //  6 MiB
  unsigned short* owb  = (unsigned short*)(ws + 23068672);   //  2 MiB
  unsigned short* qbuf = (unsigned short*)(ws + 25165824);   // 16 MiB
  unsigned short* kbuf = (unsigned short*)(ws + 41943040);   // 16 MiB
  unsigned short* vtb  = (unsigned short*)(ws + 58720256);   // 16 MiB
  unsigned short* ob   = (unsigned short*)(ws + 75497472);   // 16 MiB

  const int nx4 = (N_BATCH * N_SEQ * D_HID) / 4;   // 2,097,152
  const int nw4 = (3 * D_HID * D_HID) / 4;         //   786,432
  const int no4 = (D_HID * D_HID) / 4;             //   262,144
  cvt_bf16<<<nx4 / 256, 256, 0, stream>>>(x, xb, nx4);
  cvt_bf16<<<nw4 / 256, 256, 0, stream>>>(qkv_w, qwb, nw4);
  cvt_bf16<<<no4 / 256, 256, 0, stream>>>(o_w, owb, no4);

  // M=8192 / 256 per block; N=3072 / 64 per block
  qkv_gemm<<<dim3(32, 48), 256, 0, stream>>>(xb, qwb, qkv_b, qbuf, kbuf, vtb);

  // 32 query tiles of 64, 64 (batch,head) pairs
  attn<<<dim3(32, 64), 128, 0, stream>>>(qbuf, kbuf, vtb, ob);

  // M=8192 / 256, N=1024 / 64
  out_gemm<<<dim3(32, 16), 256, 0, stream>>>(ob, owb, o_b, out);
}
